// FunICross_90598040142137
// MI455X (gfx1250) — compile-verified, matrix operand order checked by reading, not run on hardware
//
#include <hip/hip_runtime.h>
#include <hip/hip_bf16.h>

// ---------------------------------------------------------------------------
// CDNA5 (gfx1250) wave32 WMMA cross-attention block.
//   key_proj = key @ Wk + bk ; value_proj^T = (value @ Wv + bv)^T
//   flash attention over Lkv (WMMA QK^T + online softmax + WMMA PV)
//   attn_out @ Wo + bo ; LN ; relu(x@W1+bf1)@W2+bf2 ; LN
// All matmuls: v_wmma_f32_16x16x32_f16. All B operands are stored K-contiguous
// per output column so every fragment load vectorizes to global_load_b128.
// ---------------------------------------------------------------------------

typedef _Float16 h16 __attribute__((ext_vector_type(16)));
typedef _Float16 h2  __attribute__((ext_vector_type(2)));
typedef float    f8  __attribute__((ext_vector_type(8)));
typedef float    f2  __attribute__((ext_vector_type(2)));

#define DEV __device__ __forceinline__

static constexpr int BQ   = 8;
static constexpr int LQ   = 2048;
static constexpr int LKV  = 4096;
static constexpr int DIM  = 256;
static constexpr int FF   = 1024;

DEV int kpat(int v, int hi) {            // 16-bit fragment K index pattern (ISA 7.12.2)
  return 2 * v + ((v >= 4) ? 8 : 0) + 8 * hi;
}

DEV f8 zero8() {
  f8 z;
#pragma unroll
  for (int i = 0; i < 8; ++i) z[i] = 0.0f;
  return z;
}

DEV f8 wmma_f16(h16 a, h16 b, f8 c) {
  return __builtin_amdgcn_wmma_f32_16x16x32_f16(false, a, false, b, (short)0, c,
                                                false, false);
}

// A fragment (16x32, MxK) from row-major f32 source (converted inline to f16).
DEV h16 load_a_frag_f32(const float* A, int lda, int k0) {
  int lane = threadIdx.x & 31, r = lane & 15, hi = lane >> 4;
  const float* base = A + (size_t)r * lda + k0;
  h16 f;
#pragma unroll
  for (int v = 0; v < 8; ++v) {
    int k = kpat(v, hi);
    f2 t = *(const f2*)(base + k);
    f[2 * v] = (_Float16)t[0];
    f[2 * v + 1] = (_Float16)t[1];
  }
  return f;
}

// A fragment (16x32) from row-major f16 source. Vectorizes to 2x b128.
DEV h16 load_a_frag_f16(const _Float16* A, int lda, int k0) {
  int lane = threadIdx.x & 31, r = lane & 15, hi = lane >> 4;
  const _Float16* base = A + (size_t)r * lda + k0;
  h16 f;
#pragma unroll
  for (int v = 0; v < 8; ++v) {
    int k = kpat(v, hi);
    h2 t = *(const h2*)(base + k);
    f[2 * v] = t[0];
    f[2 * v + 1] = t[1];
  }
  return f;
}

// B fragment (32x16, KxN) where the source is K-contiguous per column:
// f(k,n) = S[(row0+n)*ld + k0+k]. Vectorizes to 2x global_load_b128.
DEV h16 load_bT_frag(const _Float16* S, int ld, int row0, int k0) {
  int lane = threadIdx.x & 31, n = lane & 15, hi = lane >> 4;
  const _Float16* base = S + (size_t)(row0 + n) * ld + k0;
  h16 f;
#pragma unroll
  for (int v = 0; v < 8; ++v) {
    int k = kpat(v, hi);
    h2 t = *(const h2*)(base + k);
    f[2 * v] = t[0];
    f[2 * v + 1] = t[1];
  }
  return f;
}

// ---------------------------------------------------------------------------
// Generic WMMA GEMM: C[M,N] = act(A[M,K] @ B + bias), B given TRANSPOSED as
// Bt[N,K] (K-contiguous). Block = 8 waves, wave = 16x16 tile;
// grid (M/128, N/16). OUT_TRANS stores C^T per batch: [M/LKV, N, LKV].
// ---------------------------------------------------------------------------
template <bool A_F32, bool RELU, bool OUT_F16, bool OUT_TRANS>
__global__ void __launch_bounds__(256)
gemm_kernel(const void* __restrict__ Av, const _Float16* __restrict__ Bt,
            const float* __restrict__ bias, void* __restrict__ Cv,
            int M, int N, int K) {
  int wave = threadIdx.x >> 5, lane = threadIdx.x & 31;
  int m0 = blockIdx.x * 128 + wave * 16;
  int n0 = blockIdx.y * 16;
  f8 acc = zero8();
#pragma unroll 4
  for (int k0 = 0; k0 < K; k0 += 32) {
    h16 a;
    if constexpr (A_F32)
      a = load_a_frag_f32((const float*)Av + (size_t)m0 * K, K, k0);
    else
      a = load_a_frag_f16((const _Float16*)Av + (size_t)m0 * K, K, k0);
    h16 b = load_bT_frag(Bt, K, n0, k0);
    acc = wmma_f16(a, b, acc);
  }
  int n = lane & 15, hi = lane >> 4;
  float bv = bias ? bias[n0 + n] : 0.0f;
#pragma unroll
  for (int r = 0; r < 8; ++r) {
    float val = acc[r] + bv;
    if constexpr (RELU) val = fmaxf(val, 0.0f);
    int m = m0 + hi * 8 + r;
    if constexpr (OUT_TRANS) {
      // C^T per batch of LKV rows: out[b][n][kv], for the V projection.
      int bi = m >> 12;            // m / LKV
      int kv = m & (LKV - 1);
      size_t idx = (size_t)bi * DIM * LKV + (size_t)(n0 + n) * LKV + kv;
      ((_Float16*)Cv)[idx] = (_Float16)val;
    } else {
      size_t idx = (size_t)m * N + n0 + n;
      if constexpr (OUT_F16)
        ((_Float16*)Cv)[idx] = (_Float16)val;
      else
        ((float*)Cv)[idx] = val;
    }
  }
}

// ---------------------------------------------------------------------------
// Flash attention: grid (LQ/128, B), 8 waves/block, wave = 16 Q rows.
// Per 32-wide KV tile:
//   preload 16 K fragments (32 b128 loads in one clause) -> 16 S WMMAs
//   preload 16 V fragments (latency hidden under softmax VALU + LDS P bounce)
//   online softmax in registers -> 16 PV WMMAs.
// ---------------------------------------------------------------------------
__global__ void __launch_bounds__(256)
flash_kernel(const float* __restrict__ Q, const _Float16* __restrict__ Kp,
             const _Float16* __restrict__ VpT, _Float16* __restrict__ ctx) {
  __shared__ __align__(16) _Float16 pbuf[8][16 * 32];
  int wave = threadIdx.x >> 5, lane = threadIdx.x & 31;
  int n = lane & 15, hi = lane >> 4;
  size_t qrow0 = (size_t)blockIdx.y * LQ + blockIdx.x * 128 + wave * 16;
  const float* Qb = Q + qrow0 * DIM;
  const _Float16* Kb = Kp + (size_t)blockIdx.y * LKV * DIM;   // [kv][d]
  const _Float16* Vt = VpT + (size_t)blockIdx.y * DIM * LKV;  // [d][kv]

  // Q resident in registers: 8 A-fragments covering K = 0..255.
  h16 qf[8];
#pragma unroll
  for (int kc = 0; kc < 8; ++kc) qf[kc] = load_a_frag_f32(Qb, DIM, kc * 32);

  float m_i[8], l_i[8];
#pragma unroll
  for (int r = 0; r < 8; ++r) { m_i[r] = -1e30f; l_i[r] = 0.0f; }
  f8 acc[16];
#pragma unroll
  for (int d = 0; d < 16; ++d) acc[d] = zero8();

  const float sscale = 0.0625f;  // 1/sqrt(256)

  for (int kv0 = 0; kv0 < LKV; kv0 += 32) {
    if (kv0 + 32 < LKV) {  // CDNA5 global prefetch on the streamed KV operands
      __builtin_prefetch(Kb + (size_t)(kv0 + 32 + n) * DIM, 0, 1);
      __builtin_prefetch(Vt + (size_t)(n * 16) * LKV + kv0 + 32, 0, 1);
    }

    // ---- preload all K^T fragments for this tile, then dense WMMA chain ----
    h16 kb[16];
#pragma unroll
    for (int kc = 0; kc < 8; ++kc) {
      kb[kc]     = load_bT_frag(Kb, DIM, kv0,      kc * 32);
      kb[8 + kc] = load_bT_frag(Kb, DIM, kv0 + 16, kc * 32);
    }
    f8 s0 = zero8(), s1 = zero8();
#pragma unroll
    for (int kc = 0; kc < 8; ++kc) s0 = wmma_f16(qf[kc], kb[kc], s0);
#pragma unroll
    for (int kc = 0; kc < 8; ++kc) s1 = wmma_f16(qf[kc], kb[8 + kc], s1);

    // ---- issue V fragment loads now; latency hides under softmax ----------
    h16 vb[16];
#pragma unroll
    for (int d = 0; d < 16; ++d)
      vb[d] = load_bT_frag(Vt, LKV, d * 16, kv0);  // f(k,dd)=V[kv0+k][d*16+dd]

    // ---- online softmax (row stats live in a fixed 16-lane half-wave) -----
    float p0[8], p1[8], alpha[8];
#pragma unroll
    for (int r = 0; r < 8; ++r) {
      float a = s0[r] * sscale, b = s1[r] * sscale;
      float t = fmaxf(a, b);
      t = fmaxf(t, __shfl_xor(t, 1, 32));
      t = fmaxf(t, __shfl_xor(t, 2, 32));
      t = fmaxf(t, __shfl_xor(t, 4, 32));
      t = fmaxf(t, __shfl_xor(t, 8, 32));
      float mn = fmaxf(m_i[r], t);
      float al = __expf(m_i[r] - mn);
      float e0 = __expf(a - mn), e1 = __expf(b - mn);
      float rs = e0 + e1;
      rs += __shfl_xor(rs, 1, 32);
      rs += __shfl_xor(rs, 2, 32);
      rs += __shfl_xor(rs, 4, 32);
      rs += __shfl_xor(rs, 8, 32);
      l_i[r] = l_i[r] * al + rs;
      m_i[r] = mn;
      alpha[r] = al;
      p0[r] = e0;
      p1[r] = e1;
    }
#pragma unroll
    for (int d = 0; d < 16; ++d)
#pragma unroll
      for (int r = 0; r < 8; ++r) acc[d][r] *= alpha[r];

    // ---- C-layout -> A-layout relayout of P via this wave's LDS tile ------
    _Float16* pb = pbuf[wave];
#pragma unroll
    for (int r = 0; r < 8; ++r) {
      int row = hi * 8 + r;
      pb[row * 32 + n] = (_Float16)p0[r];
      pb[row * 32 + 16 + n] = (_Float16)p1[r];
    }
    asm volatile("s_wait_dscnt 0" ::: "memory");   // cross-lane LDS RAW
    __builtin_amdgcn_wave_barrier();
    h16 pf;
    {
      int r2 = lane & 15, h2i = lane >> 4;
#pragma unroll
      for (int v = 0; v < 8; ++v) {
        int k = kpat(v, h2i);
        h2 t = *(const h2*)(pb + r2 * 32 + k);
        pf[2 * v] = t[0];
        pf[2 * v + 1] = t[1];
      }
    }

    // ---- acc[d] += P (16x32) @ V tile (32x16), dense WMMA chain -----------
#pragma unroll
    for (int d = 0; d < 16; ++d) acc[d] = wmma_f16(pf, vb[d], acc[d]);
  }

  // Normalize and store context (f16 feeds the Wo GEMM).
#pragma unroll
  for (int r = 0; r < 8; ++r) {
    float inv = 1.0f / l_i[r];
    size_t row = qrow0 + hi * 8 + r;
#pragma unroll
    for (int d = 0; d < 16; ++d)
      ctx[row * DIM + d * 16 + n] = (_Float16)(acc[d][r] * inv);
  }
}

// ---------------------------------------------------------------------------
// out = LayerNorm(a + b) * gamma + beta, one 256-thread block per row (D=256).
// ---------------------------------------------------------------------------
__global__ void __launch_bounds__(256)
add_ln_kernel(const float* __restrict__ a, const float* __restrict__ b,
              const float* __restrict__ gamma, const float* __restrict__ beta,
              float* __restrict__ out32, _Float16* __restrict__ outh) {
  __shared__ float rs[8], rq[8];
  int tid = threadIdx.x;
  size_t idx = (size_t)blockIdx.x * DIM + tid;
  float v = a[idx] + b[idx];
  float s = v, q = v * v;
#pragma unroll
  for (int off = 16; off >= 1; off >>= 1) {
    s += __shfl_xor(s, off, 32);
    q += __shfl_xor(q, off, 32);
  }
  if ((tid & 31) == 0) { rs[tid >> 5] = s; rq[tid >> 5] = q; }
  __syncthreads();
  float tot = 0.0f, totq = 0.0f;
#pragma unroll
  for (int i = 0; i < 8; ++i) { tot += rs[i]; totq += rq[i]; }
  float mean = tot * (1.0f / 256.0f);
  float var = totq * (1.0f / 256.0f) - mean * mean;
  float rstd = rsqrtf(var + 1e-5f);
  float xn = (v - mean) * rstd * gamma[tid] + beta[tid];
  if (out32) out32[idx] = xn;
  if (outh) outh[idx] = (_Float16)xn;
}

// Convert W[K,N] f32 -> Wt[N,K] f16 (so GEMM B fragments load as b128).
__global__ void __launch_bounds__(256)
cvt_transpose_kernel(const float* __restrict__ in, _Float16* __restrict__ out,
                     int K, int N) {
  int i = blockIdx.x * 256 + threadIdx.x;
  if (i < K * N) {
    int k = i / N, n = i - k * N;
    out[(size_t)n * K + k] = (_Float16)in[i];
  }
}

// ---------------------------------------------------------------------------
extern "C" void kernel_launch(void* const* d_in, const int* in_sizes, int n_in,
                              void* d_out, int out_size, void* d_ws,
                              size_t ws_size, hipStream_t stream) {
  (void)in_sizes; (void)n_in; (void)out_size; (void)ws_size;
  const float* query = (const float*)d_in[0];
  const float* key   = (const float*)d_in[1];
  const float* value = (const float*)d_in[2];
  const float* Wk = (const float*)d_in[3];
  const float* bk = (const float*)d_in[4];
  const float* Wv = (const float*)d_in[5];
  const float* bv = (const float*)d_in[6];
  const float* Wo = (const float*)d_in[7];
  const float* bo = (const float*)d_in[8];
  const float* g1 = (const float*)d_in[9];
  const float* b1 = (const float*)d_in[10];
  const float* g2 = (const float*)d_in[11];
  const float* b2 = (const float*)d_in[12];
  const float* W1 = (const float*)d_in[13];
  const float* bf1 = (const float*)d_in[14];
  const float* W2 = (const float*)d_in[15];
  const float* bf2 = (const float*)d_in[16];
  float* out = (float*)d_out;

  // Workspace layout (lifetime-based aliasing, ~74 MB total).
  char* ws = (char*)d_ws;
  _Float16* Wk_t = (_Float16*)(ws + 0);                    // 128 KB [N,K]
  _Float16* Wv_t = (_Float16*)(ws + 131072);               // 128 KB
  _Float16* Wo_t = (_Float16*)(ws + 262144);               // 128 KB
  _Float16* W1_t = (_Float16*)(ws + 393216);               // 512 KB [FF,DIM]
  _Float16* W2_t = (_Float16*)(ws + 917504);               // 512 KB [DIM,FF]
  _Float16* Kp_h = (_Float16*)(ws + 2097152);              // 16 MB [B,LKV,DIM]
  _Float16* VpT  = (_Float16*)(ws + 18874368);             // 16 MB [B,DIM,LKV]
  _Float16* Hf_h = Kp_h;             // FFN hidden (32 MB) reuses Kp+Vp region
  _Float16* ctx_h = (_Float16*)(ws + 35651584);            // 8 MB
  _Float16* x_h = ctx_h;             // x (f16) reuses ctx after Wo GEMM
  float* ybuf = (float*)(ws + 44040192);                   // 16 MB (attn_out, then z)
  float* xbuf = (float*)(ws + 60817408);                   // 16 MB (f32 residual)

  const int Mtok = BQ * LQ;   // 16384
  const int Mkv  = BQ * LKV;  // 32768

  // 1) weights -> transposed f16
  cvt_transpose_kernel<<<DIM * DIM / 256, 256, 0, stream>>>(Wk, Wk_t, DIM, DIM);
  cvt_transpose_kernel<<<DIM * DIM / 256, 256, 0, stream>>>(Wv, Wv_t, DIM, DIM);
  cvt_transpose_kernel<<<DIM * DIM / 256, 256, 0, stream>>>(Wo, Wo_t, DIM, DIM);
  cvt_transpose_kernel<<<DIM * FF / 256, 256, 0, stream>>>(W1, W1_t, DIM, FF);
  cvt_transpose_kernel<<<DIM * FF / 256, 256, 0, stream>>>(W2, W2_t, FF, DIM);

  // 2) K projection (row-major f16 out) / V projection (transposed f16 out)
  gemm_kernel<true, false, true, false>
      <<<dim3(Mkv / 128, DIM / 16), 256, 0, stream>>>(key, Wk_t, bk, Kp_h,
                                                      Mkv, DIM, DIM);
  gemm_kernel<true, false, true, true>
      <<<dim3(Mkv / 128, DIM / 16), 256, 0, stream>>>(value, Wv_t, bv, VpT,
                                                      Mkv, DIM, DIM);

  // 3) flash attention -> ctx (f16)
  flash_kernel<<<dim3(LQ / 128, BQ), 256, 0, stream>>>(query, Kp_h, VpT, ctx_h);

  // 4) output projection -> ybuf (f32)
  gemm_kernel<false, false, false, false>
      <<<dim3(Mtok / 128, DIM / 16), 256, 0, stream>>>(ctx_h, Wo_t, bo, ybuf,
                                                       Mtok, DIM, DIM);

  // 5) x = LN(query + attn_out): f32 residual + f16 copy
  add_ln_kernel<<<Mtok, 256, 0, stream>>>(query, ybuf, g1, b1, xbuf, x_h);

  // 6) FFN: h = relu(x @ W1 + bf1)  (f16 out)
  gemm_kernel<false, true, true, false>
      <<<dim3(Mtok / 128, FF / 16), 256, 0, stream>>>(x_h, W1_t, bf1, Hf_h,
                                                      Mtok, FF, DIM);

  // 7) z = h @ W2 + bf2  (f32, reuses ybuf)
  gemm_kernel<false, false, false, false>
      <<<dim3(Mtok / 128, DIM / 16), 256, 0, stream>>>(Hf_h, W2_t, bf2, ybuf,
                                                       Mtok, DIM, FF);

  // 8) out = LN(x + z)
  add_ln_kernel<<<Mtok, 256, 0, stream>>>(xbuf, ybuf, g2, b2, out, nullptr);
}
